// MultipleHeadAttention_36876589204304
// MI455X (gfx1250) — compile-verified, matrix operand order checked by reading, not run on hardware
//
#include <hip/hip_runtime.h>
#include <hip/hip_bf16.h>

#define EMBED 1024
#define HEADS 16
#define DEPTH 64
#define BATCH 4
#define SEQ   2048

typedef __bf16 bf16_t;
typedef __attribute__((ext_vector_type(16))) bf16_t v16bf;
typedef __attribute__((ext_vector_type(8)))  bf16_t v8bf;
typedef __attribute__((ext_vector_type(8)))  float  v8f;

// ---------------------------------------------------------------------------
// Fragment loader. Per CDNA5 ISA 16-bit operand layout:
//   lane L holds row/col (L & 15); element e maps to K = (e<8?e:e+8) + 8*(L>>4)
// i.e. two contiguous 16-byte chunks at k-offsets {base, base+16}.
// Works for A (row-major M x K tile) and B (store B^T row-major: N x K tile).
// ---------------------------------------------------------------------------
__device__ __forceinline__ v16bf load_frag(const bf16_t* base, int stride) {
  const int lane = threadIdx.x & 31;
  const int row  = lane & 15;
  const int kb   = (lane >> 4) << 3;
  const bf16_t* p = base + (size_t)row * stride + kb;
  v8bf lo = *(const v8bf*)(p);
  v8bf hi = *(const v8bf*)(p + 16);
  v16bf r;
#pragma unroll
  for (int i = 0; i < 8; ++i) { r[i] = lo[i]; r[i + 8] = hi[i]; }
  return r;
}

__device__ __forceinline__ v8f wmma_bf16(v16bf a, v16bf b, v8f c) {
  return __builtin_amdgcn_wmma_f32_16x16x32_bf16(false, a, false, b, (short)0, c,
                                                 false, false);
}

// ---------------------------------------------------------------------------
// fp32 -> bf16 elementwise convert
// ---------------------------------------------------------------------------
__global__ void cvt_kernel(const float* __restrict__ s, bf16_t* __restrict__ d, int n) {
  int i = blockIdx.x * blockDim.x + threadIdx.x;
  if (i < n) d[i] = (bf16_t)s[i];
}

// fp32 [G,R,C] -> bf16 [G,C,R] (transpose inner dims) for contiguous B-operands.
__global__ void cvtT_kernel(const float* __restrict__ s, bf16_t* __restrict__ d,
                            int G, int R, int C) {
  int i = blockIdx.x * blockDim.x + threadIdx.x;
  int total = G * R * C;
  if (i >= total) return;
  int g   = i / (R * C);
  int rem = i - g * (R * C);
  int r   = rem / C;
  int c   = rem - r * C;
  d[(size_t)g * R * C + (size_t)c * R + r] = (bf16_t)s[i];
}

// ---------------------------------------------------------------------------
// Per-head projection, M=32 rows per wave, double-buffered K-loop.
// WT is [H, D, E] bf16. transpose_out=1 writes [bh, d, s] (V path).
// ---------------------------------------------------------------------------
__global__ void __launch_bounds__(32)
proj_kernel(const bf16_t* __restrict__ X, const bf16_t* __restrict__ WT,
            const float* __restrict__ bias, bf16_t* __restrict__ Y,
            int transpose_out) {
  const int MT = SEQ / 32;
  int mt = blockIdx.x % MT;
  int bh = blockIdx.x / MT;
  int h  = bh % HEADS;
  int b  = bh / HEADS;
  int lane = threadIdx.x & 31;
  int hf = lane >> 4, nx = lane & 15;

  const bf16_t* x0 = X + ((size_t)b * SEQ + (size_t)mt * 32) * EMBED;
  const bf16_t* x1 = x0 + (size_t)16 * EMBED;
  const bf16_t* wb = WT + (size_t)h * DEPTH * EMBED;

  v8f acc[2][4] = {};
  v16bf aA[2], wA[4], aB[2], wB[4];

  // prime buffer A with e = 0
  aA[0] = load_frag(x0, EMBED);
  aA[1] = load_frag(x1, EMBED);
#pragma unroll
  for (int n = 0; n < 4; ++n) wA[n] = load_frag(wb + (size_t)(n * 16) * EMBED, EMBED);

  for (int e = 0; e < EMBED; e += 64) {
    // in-flight loads for e+32 while buffer-A WMMAs issue
    aB[0] = load_frag(x0 + e + 32, EMBED);
    aB[1] = load_frag(x1 + e + 32, EMBED);
#pragma unroll
    for (int n = 0; n < 4; ++n)
      wB[n] = load_frag(wb + (size_t)(n * 16) * EMBED + e + 32, EMBED);
#pragma unroll
    for (int mi = 0; mi < 2; ++mi)
#pragma unroll
      for (int n = 0; n < 4; ++n)
        acc[mi][n] = wmma_bf16(aA[mi], wA[n], acc[mi][n]);

    if (e + 64 < EMBED) {
      aA[0] = load_frag(x0 + e + 64, EMBED);
      aA[1] = load_frag(x1 + e + 64, EMBED);
#pragma unroll
      for (int n = 0; n < 4; ++n)
        wA[n] = load_frag(wb + (size_t)(n * 16) * EMBED + e + 64, EMBED);
    }
#pragma unroll
    for (int mi = 0; mi < 2; ++mi)
#pragma unroll
      for (int n = 0; n < 4; ++n)
        acc[mi][n] = wmma_bf16(aB[mi], wB[n], acc[mi][n]);
  }

#pragma unroll
  for (int mi = 0; mi < 2; ++mi)
#pragma unroll
    for (int n = 0; n < 4; ++n) {
      int d = n * 16 + nx;
      float bv = bias[h * DEPTH + d];
#pragma unroll
      for (int i = 0; i < 8; ++i) {
        int s = mt * 32 + mi * 16 + i + 8 * hf;
        float v = acc[mi][n][i] + bv;
        if (transpose_out)
          Y[((size_t)bh * DEPTH + d) * SEQ + s] = (bf16_t)v;
        else
          Y[((size_t)bh * SEQ + s) * DEPTH + d] = (bf16_t)v;
      }
    }
}

// ---------------------------------------------------------------------------
// Flash attention, 32 queries per wave. K/V fragments are shared by both
// 16-row query sub-tiles (16 WMMAs per 32-key chunk). V loads are issued
// before the softmax VALU phase so exp/shfl work hides their latency.
// ---------------------------------------------------------------------------
__global__ void __launch_bounds__(32)
attn_kernel(const bf16_t* __restrict__ Qh, const bf16_t* __restrict__ Kh,
            const bf16_t* __restrict__ VT, bf16_t* __restrict__ Cc,
            const int* __restrict__ use_mask_p) {
  const int MT = SEQ / 32;
  int mt = blockIdx.x % MT;
  int bh = blockIdx.x / MT;
  int h  = bh % HEADS;
  int b  = bh / HEADS;
  int lane = threadIdx.x & 31;
  int hf = lane >> 4, nx = lane & 15;
  int qBase = mt * 32;
  int mask = *use_mask_p;

  const bf16_t* qb = Qh + ((size_t)bh * SEQ + qBase) * DEPTH;
  const bf16_t* kb = Kh + (size_t)bh * SEQ * DEPTH;
  const bf16_t* vb = VT + (size_t)bh * DEPTH * SEQ;

  v16bf aq[2][2];
#pragma unroll
  for (int mi = 0; mi < 2; ++mi) {
    aq[mi][0] = load_frag(qb + (size_t)(mi * 16) * DEPTH, DEPTH);
    aq[mi][1] = load_frag(qb + (size_t)(mi * 16) * DEPTH + 32, DEPTH);
  }

  v8f o[2][4] = {};
  float m[2][8], l[2][8];
#pragma unroll
  for (int mi = 0; mi < 2; ++mi)
#pragma unroll
    for (int i = 0; i < 8; ++i) { m[mi][i] = -3.0e38f; l[mi][i] = 0.0f; }

  __shared__ bf16_t pt[32][32];

  const float scale = 0.125f;  // 1/sqrt(64)
  int kEnd = mask ? (qBase + 32) : SEQ;

  for (int kc = 0; kc < kEnd; kc += 32) {
    if (kc + 32 < kEnd) {
      __builtin_prefetch(kb + (size_t)(kc + 32) * DEPTH, 0, 0);
      __builtin_prefetch(vb + (size_t)(kc + 32), 0, 0);
    }
    // ---- logits: 32q x 32k = four 16x16 C fragments ----
    v16bf bk[2][2];
#pragma unroll
    for (int f = 0; f < 2; ++f) {
      bk[f][0] = load_frag(kb + (size_t)(kc + f * 16) * DEPTH, DEPTH);
      bk[f][1] = load_frag(kb + (size_t)(kc + f * 16) * DEPTH + 32, DEPTH);
    }
    v8f lg[2][2] = {};
#pragma unroll
    for (int mi = 0; mi < 2; ++mi)
#pragma unroll
      for (int f = 0; f < 2; ++f) {
        lg[mi][f] = wmma_bf16(aq[mi][0], bk[f][0], lg[mi][f]);
        lg[mi][f] = wmma_bf16(aq[mi][1], bk[f][1], lg[mi][f]);
      }
    // ---- issue V loads now; softmax VALU below hides their latency ----
    v16bf bv[4];
#pragma unroll
    for (int n = 0; n < 4; ++n)
      bv[n] = load_frag(vb + (size_t)(n * 16) * SEQ + kc, SEQ);

    // ---- scale + causal mask ----
#pragma unroll
    for (int mi = 0; mi < 2; ++mi)
#pragma unroll
      for (int f = 0; f < 2; ++f)
#pragma unroll
        for (int i = 0; i < 8; ++i) {
          int kg = kc + f * 16 + nx;
          int qg = qBase + mi * 16 + i + 8 * hf;
          float v = lg[mi][f][i] * scale;
          lg[mi][f][i] = (mask && (kg > qg)) ? -1.0e30f : v;
        }
    // ---- online softmax stats ----
    float corr[2][8];
#pragma unroll
    for (int mi = 0; mi < 2; ++mi)
#pragma unroll
      for (int i = 0; i < 8; ++i) {
        float t = fmaxf(lg[mi][0][i], lg[mi][1][i]);
        t = fmaxf(t, __shfl_xor(t, 1, 32));
        t = fmaxf(t, __shfl_xor(t, 2, 32));
        t = fmaxf(t, __shfl_xor(t, 4, 32));
        t = fmaxf(t, __shfl_xor(t, 8, 32));
        float mn = fmaxf(m[mi][i], t);
        corr[mi][i] = __expf(m[mi][i] - mn);
        m[mi][i] = mn;
      }
#pragma unroll
    for (int mi = 0; mi < 2; ++mi)
#pragma unroll
      for (int i = 0; i < 8; ++i) {
        float p0 = __expf(lg[mi][0][i] - m[mi][i]);
        float p1 = __expf(lg[mi][1][i] - m[mi][i]);
        pt[mi * 16 + i + 8 * hf][nx]      = (bf16_t)p0;
        pt[mi * 16 + i + 8 * hf][16 + nx] = (bf16_t)p1;
        float s = p0 + p1;
        s += __shfl_xor(s, 1, 32);
        s += __shfl_xor(s, 2, 32);
        s += __shfl_xor(s, 4, 32);
        s += __shfl_xor(s, 8, 32);
        l[mi][i] = l[mi][i] * corr[mi][i] + s;
      }
    __syncthreads();
    v16bf ap[2];
    ap[0] = load_frag(&pt[0][0], 32);
    ap[1] = load_frag(&pt[16][0], 32);
    // ---- O = O*corr + P @ V ----
#pragma unroll
    for (int mi = 0; mi < 2; ++mi)
#pragma unroll
      for (int n = 0; n < 4; ++n) {
#pragma unroll
        for (int i = 0; i < 8; ++i) o[mi][n][i] *= corr[mi][i];
        o[mi][n] = wmma_bf16(ap[mi], bv[n], o[mi][n]);
      }
    __syncthreads();
  }
  // ---- normalize and write concat [B, S, H*D] ----
#pragma unroll
  for (int mi = 0; mi < 2; ++mi)
#pragma unroll
    for (int n = 0; n < 4; ++n)
#pragma unroll
      for (int i = 0; i < 8; ++i) {
        int s   = qBase + mi * 16 + i + 8 * hf;
        int col = h * DEPTH + n * 16 + nx;
        float v = o[mi][n][i] / l[mi][i];
        Cc[((size_t)b * SEQ + s) * EMBED + col] = (bf16_t)v;
      }
}

// ---------------------------------------------------------------------------
// Output projection: out = concat @ wo + bo. M=32 rows x N=64 cols per wave,
// double-buffered K-loop. WoT is [E_out, E_in] bf16.
// ---------------------------------------------------------------------------
__global__ void __launch_bounds__(32)
oproj_kernel(const bf16_t* __restrict__ Cc, const bf16_t* __restrict__ WoT,
             const float* __restrict__ bo, float* __restrict__ out) {
  int nt = blockIdx.x & 15;
  int mt = blockIdx.x >> 4;
  int lane = threadIdx.x & 31;
  int hf = lane >> 4, nx = lane & 15;

  const bf16_t* x0 = Cc + (size_t)mt * 32 * EMBED;
  const bf16_t* x1 = x0 + (size_t)16 * EMBED;
  const bf16_t* wb = WoT + (size_t)(nt * 64) * EMBED;

  v8f acc[2][4] = {};
  v16bf aA[2], wA[4], aB[2], wB[4];

  aA[0] = load_frag(x0, EMBED);
  aA[1] = load_frag(x1, EMBED);
#pragma unroll
  for (int n = 0; n < 4; ++n) wA[n] = load_frag(wb + (size_t)(n * 16) * EMBED, EMBED);

  for (int e = 0; e < EMBED; e += 64) {
    aB[0] = load_frag(x0 + e + 32, EMBED);
    aB[1] = load_frag(x1 + e + 32, EMBED);
#pragma unroll
    for (int n = 0; n < 4; ++n)
      wB[n] = load_frag(wb + (size_t)(n * 16) * EMBED + e + 32, EMBED);
#pragma unroll
    for (int mi = 0; mi < 2; ++mi)
#pragma unroll
      for (int n = 0; n < 4; ++n)
        acc[mi][n] = wmma_bf16(aA[mi], wA[n], acc[mi][n]);

    if (e + 64 < EMBED) {
      aA[0] = load_frag(x0 + e + 64, EMBED);
      aA[1] = load_frag(x1 + e + 64, EMBED);
#pragma unroll
      for (int n = 0; n < 4; ++n)
        wA[n] = load_frag(wb + (size_t)(n * 16) * EMBED + e + 64, EMBED);
    }
#pragma unroll
    for (int mi = 0; mi < 2; ++mi)
#pragma unroll
      for (int n = 0; n < 4; ++n)
        acc[mi][n] = wmma_bf16(aB[mi], wB[n], acc[mi][n]);
  }

#pragma unroll
  for (int mi = 0; mi < 2; ++mi)
#pragma unroll
    for (int n = 0; n < 4; ++n) {
      int col = nt * 64 + n * 16 + nx;
      float bb = bo[col];
#pragma unroll
      for (int i = 0; i < 8; ++i) {
        int row = mt * 32 + mi * 16 + i + 8 * hf;
        out[(size_t)row * EMBED + col] = acc[mi][n][i] + bb;
      }
    }
}

// ---------------------------------------------------------------------------
extern "C" void kernel_launch(void* const* d_in, const int* in_sizes, int n_in,
                              void* d_out, int out_size, void* d_ws, size_t ws_size,
                              hipStream_t stream) {
  (void)in_sizes; (void)n_in; (void)out_size; (void)ws_size;
  const float* q  = (const float*)d_in[0];
  const float* k  = (const float*)d_in[1];
  const float* v  = (const float*)d_in[2];
  const float* wq = (const float*)d_in[3];
  const float* bq = (const float*)d_in[4];
  const float* wk = (const float*)d_in[5];
  const float* bk = (const float*)d_in[6];
  const float* wv = (const float*)d_in[7];
  const float* bv = (const float*)d_in[8];
  const float* wo = (const float*)d_in[9];
  const float* bo = (const float*)d_in[10];
  const int* use_mask = (const int*)d_in[11];
  float* out = (float*)d_out;

  char* ws = (char*)d_ws;
  const size_t MB = 1ull << 20;
  bf16_t* qbf = (bf16_t*)(ws +   0 * MB);   // [B,S,E]      16 MB
  bf16_t* kbf = (bf16_t*)(ws +  16 * MB);   // [B,S,E]      16 MB
  bf16_t* vbf = (bf16_t*)(ws +  32 * MB);   // [B,S,E]      16 MB
  bf16_t* wqT = (bf16_t*)(ws +  48 * MB);   // [H,D,E]       2 MB
  bf16_t* wkT = (bf16_t*)(ws +  50 * MB);   // [H,D,E]       2 MB
  bf16_t* wvT = (bf16_t*)(ws +  52 * MB);   // [H,D,E]       2 MB
  bf16_t* woT = (bf16_t*)(ws +  54 * MB);   // [E,E]         2 MB
  bf16_t* qh  = (bf16_t*)(ws +  56 * MB);   // [BH,S,D]     16 MB
  bf16_t* kh  = (bf16_t*)(ws +  72 * MB);   // [BH,S,D]     16 MB
  bf16_t* vT  = (bf16_t*)(ws +  88 * MB);   // [BH,D,S]     16 MB
  bf16_t* cc  = (bf16_t*)(ws + 104 * MB);   // [B,S,E]      16 MB

  const int nX = BATCH * SEQ * EMBED;
  cvt_kernel<<<(nX + 255) / 256, 256, 0, stream>>>(q, qbf, nX);
  cvt_kernel<<<(nX + 255) / 256, 256, 0, stream>>>(k, kbf, nX);
  cvt_kernel<<<(nX + 255) / 256, 256, 0, stream>>>(v, vbf, nX);

  const int nW = HEADS * EMBED * DEPTH;
  cvtT_kernel<<<(nW + 255) / 256, 256, 0, stream>>>(wq, wqT, HEADS, EMBED, DEPTH);
  cvtT_kernel<<<(nW + 255) / 256, 256, 0, stream>>>(wk, wkT, HEADS, EMBED, DEPTH);
  cvtT_kernel<<<(nW + 255) / 256, 256, 0, stream>>>(wv, wvT, HEADS, EMBED, DEPTH);
  const int nWo = EMBED * EMBED;
  cvtT_kernel<<<(nWo + 255) / 256, 256, 0, stream>>>(wo, woT, 1, EMBED, EMBED);

  const int nTiles32 = BATCH * HEADS * (SEQ / 32);   // 4096 waves
  proj_kernel<<<nTiles32, 32, 0, stream>>>(qbf, wqT, bq, qh, 0);
  proj_kernel<<<nTiles32, 32, 0, stream>>>(kbf, wkT, bk, kh, 0);
  proj_kernel<<<nTiles32, 32, 0, stream>>>(vbf, wvT, bv, vT, 1);

  attn_kernel<<<nTiles32, 32, 0, stream>>>(qh, kh, vT, cc, use_mask);

  oproj_kernel<<<(BATCH * SEQ / 32) * (EMBED / 64), 32, 0, stream>>>(cc, woT, bo, out);
}
